// RNN_8933531976222
// MI455X (gfx1250) — matrix operand order repaired and checked by reading, not compiled, not run on hardware
//
#include <hip/hip_runtime.h>
#include <hip/hip_bf16.h>
#include <math.h>

typedef __bf16 bf16_t;
typedef __attribute__((ext_vector_type(16))) __bf16 v16bf;
typedef __attribute__((ext_vector_type(8)))  __bf16 v8bf;
typedef __attribute__((ext_vector_type(8)))  float  v8f;

#define AS3 __attribute__((address_space(3)))

#define BB 256
#define TT 256
#define II 512
#define HH 512

// ---------------------------------------------------------------------------
// Prep: fp32 -> bf16 conversion of x (grid-stride)
// ---------------------------------------------------------------------------
__global__ __launch_bounds__(256) void cvt_x_kernel(const float* __restrict__ x,
                                                    bf16_t* __restrict__ xbf, int n) {
    for (int i = blockIdx.x * blockDim.x + threadIdx.x; i < n;
         i += gridDim.x * blockDim.x)
        xbf[i] = (bf16_t)x[i];
}

// ---------------------------------------------------------------------------
// Prep: transpose W_ih [H,I]->WihT [I,H] bf16 and W_hh [H,H]->WhhT [H,H] bf16
// so WMMA B-fragments (lane = N column, 16 K rows strided by H) read a [K,N] layout.
// ---------------------------------------------------------------------------
__global__ __launch_bounds__(256) void prep_w_kernel(const float* __restrict__ Wih,
                                                     const float* __restrict__ Whh,
                                                     bf16_t* __restrict__ WihT,
                                                     bf16_t* __restrict__ WhhT) {
    int bi = blockIdx.x;
    if (bi < II) {
        int i = bi;
        for (int h = threadIdx.x; h < HH; h += 256)
            WihT[i * HH + h] = (bf16_t)Wih[h * II + i];
    } else {
        int k = bi - II;
        for (int n = threadIdx.x; n < HH; n += 256)
            WhhT[k * HH + n] = (bf16_t)Whh[n * HH + k];
    }
}

// async 16-byte global -> LDS copy (per active lane), tracked by ASYNCcnt.
__device__ __forceinline__ void async_copy_b128(const void* gptr, void* lptr) {
#if defined(__gfx1250__)
    unsigned lds_off = (unsigned)(unsigned long long)(AS3 char*)lptr;
    unsigned long long gaddr = (unsigned long long)gptr;
    asm volatile("global_load_async_to_lds_b128 %0, %1, off"
                 :: "v"(lds_off), "v"(gaddr) : "memory");
#else
    *(v8bf*)lptr = *(const v8bf*)gptr;
#endif
}

__device__ __forceinline__ void async_wait_all() {
#if defined(__gfx1250__)
    asm volatile("s_wait_asynccnt 0x0" ::: "memory");
#endif
}

// ---------------------------------------------------------------------------
// Phase 1: xp[M=B*T, H] = x_bf16 @ WihT + (b_ih + b_hh)
// Block: 256 threads = 8 waves. A block of 32 M-rows (32KB bf16) staged into
// LDS via global_load_async_to_lds_b128; waves 0-3 -> rows [0,16),
// waves 4-7 -> rows [16,32); each wave computes 16(M) x 128(N).
// Grid: M/32 = 2048 blocks.
// ---------------------------------------------------------------------------
__global__ __launch_bounds__(256) void gemm_xp_kernel(const bf16_t* __restrict__ xbf,
                                                      const bf16_t* __restrict__ WihT,
                                                      const float* __restrict__ b_ih,
                                                      const float* __restrict__ b_hh,
                                                      float* __restrict__ xp) {
    __shared__ bf16_t As[32 * II];                 // 32 KB
    const int lane = threadIdx.x & 31;
    const int wave = threadIdx.x >> 5;             // 0..7
    const int mt   = wave >> 2;                    // A sub-block 0/1
    const int n0   = (wave & 3) * 128;
    const int m0   = blockIdx.x * 32;
    const int koff = (lane >> 4) * 8;              // lanes 16-31: K+8

    // ---- stage A rows [m0, m0+32) into LDS asynchronously ----
    {
        const bf16_t* gsrc = xbf + (size_t)m0 * II;
        for (int ofs = threadIdx.x * 8; ofs < 32 * II; ofs += 256 * 8)
            async_copy_b128(gsrc + ofs, &As[ofs]);
        async_wait_all();
        __syncthreads();
    }

    v8f acc[8];
    #pragma unroll
    for (int q = 0; q < 8; ++q) acc[q] = (v8f){0.f,0.f,0.f,0.f,0.f,0.f,0.f,0.f};

    const bf16_t* abase = As + (mt * 16 + (lane & 15)) * II + koff;

    for (int k = 0; k < II; k += 32) {
        v8bf lo = *(const v8bf*)(abase + k);
        v8bf hi = *(const v8bf*)(abase + k + 16);
        v16bf a = __builtin_shufflevector(lo, hi, 0,1,2,3,4,5,6,7,8,9,10,11,12,13,14,15);
        const bf16_t* bp = WihT + (k + koff) * HH + n0 + (lane & 15);
        v16bf b[8];
        #pragma unroll
        for (int q = 0; q < 8; ++q)
            for (int s = 0; s < 16; ++s) b[q][s] = bp[((s & 7) + (s >> 3) * 16) * HH + q * 16];
        #pragma unroll
        for (int q = 0; q < 8; ++q)
            acc[q] = __builtin_amdgcn_wmma_f32_16x16x32_bf16(
                false, a, false, b[q], (short)0, acc[q], false, false);
    }
    // store + bias; C layout: VGPR r, lane l -> M = r + 8*(l>>4), N = l&15
    #pragma unroll
    for (int q = 0; q < 8; ++q) {
        int n = n0 + q * 16 + (lane & 15);
        float bias = b_ih[n] + b_hh[n];
        #pragma unroll
        for (int r = 0; r < 8; ++r) {
            int m = m0 + mt * 16 + r + 8 * (lane >> 4);
            xp[m * HH + n] = acc[q][r] + bias;
        }
    }
}

// ---------------------------------------------------------------------------
// Phase 2: serial recurrence. One block per 16-batch-row tile (16 blocks,
// dictated by the T=256 serial chain). 8 waves/block; wave owns 64 columns.
// h kept in LDS as bf16 (16 KB); WhhT (512 KB bf16) streamed from L2.
// xp overwritten in place with r_out = tanh(...).
// ---------------------------------------------------------------------------
__global__ __launch_bounds__(256) void rnn_rec_kernel(const bf16_t* __restrict__ WhhT,
                                                      float* xp) {
    __shared__ bf16_t hbuf[16 * HH];
    const int lane = threadIdx.x & 31;
    const int wave = threadIdx.x >> 5;
    const int b0   = blockIdx.x * 16;
    const int n0   = wave * 64;
    const int mrow = lane & 15;
    const int koff = (lane >> 4) * 8;

    for (int i = threadIdx.x; i < 16 * HH; i += 256) hbuf[i] = (bf16_t)0.0f;
    __syncthreads();

    for (int t = 0; t < TT; ++t) {
        v8f acc[4];
        #pragma unroll
        for (int q = 0; q < 4; ++q) {
            int n = n0 + q * 16 + (lane & 15);
            #pragma unroll
            for (int r = 0; r < 8; ++r) {
                int b = b0 + r + 8 * (lane >> 4);
                acc[q][r] = xp[(b * TT + t) * HH + n];
            }
            // warm caches for next step's accumulator init
            if (t + 1 < TT)
                __builtin_prefetch(&xp[((b0 + 8 * (lane >> 4)) * TT + t + 1) * HH + n], 0, 1);
        }
        // h @ WhhT
        for (int k = 0; k < HH; k += 32) {
            const bf16_t* ap = hbuf + mrow * HH + k + koff;
            v8bf lo = *(const v8bf*)(ap);
            v8bf hi = *(const v8bf*)(ap + 16);
            v16bf a = __builtin_shufflevector(lo, hi, 0,1,2,3,4,5,6,7,8,9,10,11,12,13,14,15);
            const bf16_t* bp = WhhT + (k + koff) * HH + n0 + (lane & 15);
            v16bf b[4];
            #pragma unroll
            for (int q = 0; q < 4; ++q)
                for (int s = 0; s < 16; ++s) b[q][s] = bp[((s & 7) + (s >> 3) * 16) * HH + q * 16];
            #pragma unroll
            for (int q = 0; q < 4; ++q)
                acc[q] = __builtin_amdgcn_wmma_f32_16x16x32_bf16(
                    false, a, false, b[q], (short)0, acc[q], false, false);
        }
        #pragma unroll
        for (int q = 0; q < 4; ++q)
            #pragma unroll
            for (int r = 0; r < 8; ++r)
                acc[q][r] = tanhf(acc[q][r]);

        __syncthreads();  // all waves done reading previous h
        #pragma unroll
        for (int q = 0; q < 4; ++q) {
            int n = n0 + q * 16 + (lane & 15);
            #pragma unroll
            for (int r = 0; r < 8; ++r) {
                int brow = r + 8 * (lane >> 4);
                hbuf[brow * HH + n] = (bf16_t)acc[q][r];
                xp[((b0 + brow) * TT + t) * HH + n] = acc[q][r];  // r_out
            }
        }
        __syncthreads();  // h visible for next step
    }
}

// ---------------------------------------------------------------------------
// Phase 3: out[m] = dot(r_out[m,:], W_out) + b_out ; one wave per output.
// ---------------------------------------------------------------------------
__global__ __launch_bounds__(256) void out_proj_kernel(const float* __restrict__ rout,
                                                       const float* __restrict__ Wout,
                                                       const float* __restrict__ bout,
                                                       float* __restrict__ out) {
    int wid  = blockIdx.x * 8 + (threadIdx.x >> 5);   // 0..65535
    int lane = threadIdx.x & 31;
    const float* row = rout + (size_t)wid * HH;
    float s = 0.f;
    for (int j = 0; j < HH; j += 32) s += row[j + lane] * Wout[j + lane];
    for (int off = 16; off > 0; off >>= 1) s += __shfl_down(s, off, 32);
    if (lane == 0) out[wid] = s + bout[0];
}

// ---------------------------------------------------------------------------
extern "C" void kernel_launch(void* const* d_in, const int* in_sizes, int n_in,
                              void* d_out, int out_size, void* d_ws, size_t ws_size,
                              hipStream_t stream) {
    (void)in_sizes; (void)n_in; (void)out_size; (void)ws_size;
    const float* x     = (const float*)d_in[0];
    const float* W_ih  = (const float*)d_in[1];
    const float* W_hh  = (const float*)d_in[2];
    const float* b_ih  = (const float*)d_in[3];
    const float* b_hh  = (const float*)d_in[4];
    const float* W_out = (const float*)d_in[5];
    const float* b_out = (const float*)d_in[6];
    float* out = (float*)d_out;

    char* ws = (char*)d_ws;
    const size_t nX = (size_t)BB * TT * II;              // 33,554,432
    bf16_t* xbf  = (bf16_t*)(ws);                        // 64 MB
    bf16_t* WihT = (bf16_t*)(ws + nX * 2);               // 512 KB
    bf16_t* WhhT = (bf16_t*)(ws + nX * 2 + II * HH * 2); // 512 KB
    float*  xp   = (float*) (ws + nX * 2 + 2 * II * HH * 2);

    cvt_x_kernel<<<4096, 256, 0, stream>>>(x, xbf, (int)nX);
    prep_w_kernel<<<2 * II, 256, 0, stream>>>(W_ih, W_hh, WihT, WhhT);
    gemm_xp_kernel<<<(BB * TT) / 32, 256, 0, stream>>>(xbf, WihT, b_ih, b_hh, xp);
    rnn_rec_kernel<<<BB / 16, 256, 0, stream>>>(WhhT, xp);
    out_proj_kernel<<<(BB * TT) / 8, 256, 0, stream>>>(xp, W_out, b_out, out);
}
